// AttentionBlock_18133351924519
// MI455X (gfx1250) — matrix operand order repaired and checked by reading, not compiled, not run on hardware
//
#include <hip/hip_runtime.h>

#define B_   2
#define C_   256
#define N_   8192      // 8*32*32
#define CQK_ 32

typedef __bf16 bf16;
typedef __attribute__((ext_vector_type(16))) __bf16 v16bf;
typedef __attribute__((ext_vector_type(8)))  float  v8f;
typedef __attribute__((ext_vector_type(4)))  unsigned int u32x4;

// ---------------------------------------------------------------------------
// Projection kernel: q = wq x + bq, k = wk x + bk, v = wv x + bv
//   qt/kt stored token-major (B, N, 32)  -> feeds WMMA operand loads
//   vc    stored channel-major (B, 256, N) -> feeds P*V staging
// ~3 GFLOP total; weights are wave-uniform (scalar loads), x tiled in LDS.
// ---------------------------------------------------------------------------
__global__ __launch_bounds__(256) void proj_qkv(
    const float* __restrict__ x,
    const float* __restrict__ wq, const float* __restrict__ bq,
    const float* __restrict__ wk, const float* __restrict__ bk,
    const float* __restrict__ wv, const float* __restrict__ bv,
    unsigned short* __restrict__ qtw, unsigned short* __restrict__ ktw,
    unsigned short* __restrict__ vcw)
{
    __shared__ float xs[C_][32];                 // 32 KB tile: all channels x 32 tokens
    const int b  = blockIdx.x / (N_ / 32);
    const int n0 = (blockIdx.x % (N_ / 32)) * 32;
    const int t  = threadIdx.x;

    {   // cooperative load: 8 threads per channel row, float4 each
        const int c = t >> 3;
        const int j = (t & 7) << 2;
        const float* src = x + ((size_t)b * C_ + c) * N_ + n0 + j;
        #pragma unroll
        for (int i = 0; i < 8; ++i)
            *(float4*)&xs[c + i * 32][j] = *(const float4*)(src + (size_t)i * 32 * N_);
    }
    __syncthreads();

    bf16* qt = reinterpret_cast<bf16*>(qtw);
    bf16* kt = reinterpret_cast<bf16*>(ktw);
    bf16* vc = reinterpret_cast<bf16*>(vcw);

    const int n = t & 31;     // token within tile (lane id -> conflict-free LDS)
    const int g = t >> 5;     // wave id -> output row group (uniform per wave)

    // V: each thread produces 32 channels for its token
    for (int i = 0; i < 32; ++i) {
        const int o = g + 8 * i;
        float a = bv[o];
        const float* wr = wv + o * C_;
        #pragma unroll 8
        for (int c2 = 0; c2 < C_; ++c2) a = fmaf(wr[c2], xs[c2][n], a);
        vc[((size_t)b * C_ + o) * N_ + n0 + n] = (bf16)a;
    }
    // Q, K: 4 outputs each
    for (int i = 0; i < 4; ++i) {
        const int o = g + 8 * i;
        float aq = bq[o], ak = bk[o];
        const float* wqr = wq + o * C_;
        const float* wkr = wk + o * C_;
        #pragma unroll 8
        for (int c2 = 0; c2 < C_; ++c2) {
            const float xv = xs[c2][n];
            aq = fmaf(wqr[c2], xv, aq);
            ak = fmaf(wkr[c2], xv, ak);
        }
        qt[((size_t)b * N_ + n0 + n) * CQK_ + o] = (bf16)aq;
        kt[((size_t)b * N_ + n0 + n) * CQK_ + o] = (bf16)ak;
    }
}

// ---------------------------------------------------------------------------
// Flash attention, LDS-staged:
//   - 8 waves / block, each wave owns a private 16-query tile (all 256 ch)
//   - K/V chunk (32 keys) staged once per block into double-buffered LDS via
//     global_load_async_to_lds_* (ASYNCcnt), cutting L2 traffic 8x
//   - Scores computed TRANSPOSED (S^T = K_tile x Q) so softmax probs land
//     exactly in the A-matrix layout of the P x V WMMA: no lane transposes
//   - 18 v_wmma_f32_16x16x32_bf16 per chunk per wave, 256 chunks
// LDS rows padded to 80 B: 16B-aligned for ds_load_b128 and bank-conflict-free
// (gcd(20 words, 64 banks) = 4 -> 16 distinct banks across each 16-lane half).
// ---------------------------------------------------------------------------
#define KROW 80                    // bytes per K row (64 data + 16 pad)
#define VROW 80                    // bytes per V row (64 data + 16 pad)

__global__ __launch_bounds__(256, 1) void flash_attn(
    const unsigned short* __restrict__ qtw, const unsigned short* __restrict__ ktw,
    const unsigned short* __restrict__ vcw, const float* __restrict__ x,
    const float* __restrict__ gammap, float* __restrict__ out)
{
    __shared__ __align__(16) unsigned char kbuf[2][32 * KROW];    //  5 KB
    __shared__ __align__(16) unsigned char vbuf[2][256 * VROW];   // 40 KB

    const int t    = threadIdx.x;
    const int lane = t & 31;
    const int wv_  = t >> 5;                     // wave id 0..7
    const int b    = blockIdx.x >> 6;            // 64 blocks per batch
    const int q0   = (((blockIdx.x & 63) << 3) + wv_) << 4;
    const int hw   = lane >> 4;                  // half-wave 0/1
    const int lq   = lane & 15;

    const bf16* qt = reinterpret_cast<const bf16*>(qtw) + (size_t)b * N_ * CQK_;
    const bf16* kt = reinterpret_cast<const bf16*>(ktw) + (size_t)b * N_ * CQK_;
    const bf16* vc = reinterpret_cast<const bf16*>(vcw) + (size_t)b * C_ * N_;

    // --- async staging assignment: every thread = 1 K b64 + 4 V b128 / chunk,
    //     so every wave issues exactly 5 async ops per chunk (uniform ASYNCcnt)
    const bf16*    kg = kt + (size_t)(t >> 3) * CQK_ + ((t & 7) << 2); // key row t/8, 8B col t%8
    const unsigned kl = (unsigned)(t >> 3) * KROW + ((t & 7) << 3);
    const bf16*    vg = vc + (size_t)t * N_;                           // channel row t
    const unsigned vl = (unsigned)t * VROW;

    const unsigned kbase0 = (unsigned)(size_t)&kbuf[0][0];
    const unsigned kbase1 = (unsigned)(size_t)&kbuf[1][0];
    const unsigned vbase0 = (unsigned)(size_t)&vbuf[0][0];
    const unsigned vbase1 = (unsigned)(size_t)&vbuf[1][0];

    #define ISSUE_CHUNK(kc, nb)                                                     \
        do {                                                                        \
            unsigned kb_ = ((nb) ? kbase1 : kbase0) + kl;                           \
            const void* kga_ = (const void*)(kg + (size_t)(kc) * CQK_);             \
            asm volatile("global_load_async_to_lds_b64 %0, %1, off"                 \
                         :: "v"(kb_), "v"(kga_) : "memory");                        \
            unsigned vb_ = ((nb) ? vbase1 : vbase0) + vl;                           \
            const bf16* vga_ = vg + (kc);                                           \
            _Pragma("unroll")                                                       \
            for (int j_ = 0; j_ < 4; ++j_) {                                        \
                unsigned lo_ = vb_ + j_ * 16;                                       \
                const void* ga_ = (const void*)(vga_ + j_ * 8);                     \
                asm volatile("global_load_async_to_lds_b128 %0, %1, off"            \
                             :: "v"(lo_), "v"(ga_) : "memory");                     \
            }                                                                       \
        } while (0)

    // Q as B-operand (32 feats x 16 queries): lane = query lq, feats hw*16..+15
    const v16bf qb = *(const v16bf*)(qt + (size_t)(q0 + lq) * CQK_ + (hw << 4));

    v8f acc[16];                                 // 16 channel tiles x 8 VGPRs
    const v8f zacc = {0.f, 0.f, 0.f, 0.f, 0.f, 0.f, 0.f, 0.f};
    #pragma unroll
    for (int i = 0; i < 16; ++i) acc[i] = zacc;

    float mrun = -__builtin_inff();              // running max (per query = lane%16)
    float lrun = 0.f;                            // running denom
    const int f0b = hw << 5;                     // byte offset of feature group {0,16}

    ISSUE_CHUNK(0, 0);

    const int NC = N_ / 32;
    for (int i = 0; i < NC; ++i) {
        const int cb = i & 1;
        if (i + 1 < NC) {
            ISSUE_CHUNK((i + 1) * 32, cb ^ 1);
            asm volatile("s_wait_asynccnt 0x5" ::: "memory");   // chunk i landed
        } else {
            asm volatile("s_wait_asynccnt 0x0" ::: "memory");
        }
        __syncthreads();                          // all waves' stages visible

        // K chunk as two A-operands (16 keys x 32 feats each) from LDS
        const unsigned char* kb = &kbuf[cb][0];
        union { v16bf v; u32x4 u[2]; } ka0, ka1;
        ka0.u[0] = *(const u32x4*)(kb + lq * KROW + f0b);
        ka0.u[1] = *(const u32x4*)(kb + lq * KROW + f0b + 16);
        ka1.u[0] = *(const u32x4*)(kb + (16 + lq) * KROW + f0b);
        ka1.u[1] = *(const u32x4*)(kb + (16 + lq) * KROW + f0b + 16);

        // S^T[key][query]: my lane holds 16 of 32 scores for query lq;
        // partner lane (^16) holds the other 16.
        v8f s0 = __builtin_amdgcn_wmma_f32_16x16x32_bf16(false, ka0.v, false, qb,
                                                         (short)0, zacc, false, false);
        v8f s1 = __builtin_amdgcn_wmma_f32_16x16x32_bf16(false, ka1.v, false, qb,
                                                         (short)0, zacc, false, false);

        // --- online softmax (one xor-16 shuffle per reduction) ---
        float smax = fmaxf(s0[0], s1[0]);
        #pragma unroll
        for (int k = 1; k < 8; ++k) smax = fmaxf(smax, fmaxf(s0[k], s1[k]));
        smax = fmaxf(smax, __shfl_xor(smax, 16, 32));
        const float mnew = fmaxf(mrun, smax);
        const float corr = __expf(mrun - mnew);   // 0 on first chunk (mrun = -inf)

        float p0[8], p1[8];
        float psum = 0.f;
        #pragma unroll
        for (int k = 0; k < 8; ++k) {
            p0[k] = __expf(s0[k] - mnew);
            p1[k] = __expf(s1[k] - mnew);
            psum += p0[k] + p1[k];
        }
        psum += __shfl_xor(psum, 16, 32);
        lrun = fmaf(lrun, corr, psum);
        mrun = mnew;

        // P packs directly into the 16x32 A-matrix layout (no lane movement)
        v16bf ap;
        #pragma unroll
        for (int k = 0; k < 8; ++k) { ap[k] = (bf16)p0[k]; ap[k + 8] = (bf16)p1[k]; }

        // broadcast rescale factor to accumulator row layout (row = v + 8*hw)
        v8f rs;
        #pragma unroll
        for (int v = 0; v < 8; ++v) rs[v] = __shfl(corr, (hw << 3) + v, 32);

        // O += P x V over 16 channel tiles; V B-operand = 32B from LDS
        const unsigned char* vbp = &vbuf[cb][0] + (size_t)lq * VROW + (hw << 4);
        #pragma unroll
        for (int ct = 0; ct < 16; ++ct) {
            union { v16bf v; u32x4 u[2]; } vbu;
            vbu.u[0] = *(const u32x4*)(vbp + ct * (16 * VROW));
            vbu.u[1] = *(const u32x4*)(vbp + ct * (16 * VROW) + 32);
            const v16bf vb = vbu.v;
            acc[ct] = __builtin_amdgcn_wmma_f32_16x16x32_bf16(
                false, ap, false, vb, (short)0, acc[ct] * rs, false, false);
        }
        __syncthreads();                          // release buffer cb for re-stage
    }

    // epilogue: O/l, then gamma*out + x with the reference's raw (N,C) flattening
    const float gamma = gammap[0];
    const float linv  = 1.f / lrun;
    v8f li;
    #pragma unroll
    for (int v = 0; v < 8; ++v) li[v] = __shfl(linv, (hw << 3) + v, 32);

    const float* xb = x + (size_t)b * C_ * N_;
    float*       ob = out + (size_t)b * C_ * N_;
    #pragma unroll
    for (int ct = 0; ct < 16; ++ct) {
        #pragma unroll
        for (int v = 0; v < 8; ++v) {
            const int    qrow = q0 + v + (hw << 3);
            const int    c    = ct * 16 + lq;
            const size_t off  = (size_t)qrow * C_ + c;
            ob[off] = fmaf(gamma, acc[ct][v] * li[v], xb[off]);
        }
    }
    #undef ISSUE_CHUNK
}

// ---------------------------------------------------------------------------
extern "C" void kernel_launch(void* const* d_in, const int* in_sizes, int n_in,
                              void* d_out, int out_size, void* d_ws, size_t ws_size,
                              hipStream_t stream)
{
    const float* x     = (const float*)d_in[0];
    const float* wq    = (const float*)d_in[1];
    const float* bq    = (const float*)d_in[2];
    const float* wk    = (const float*)d_in[3];
    const float* bk    = (const float*)d_in[4];
    const float* wv    = (const float*)d_in[5];
    const float* bv    = (const float*)d_in[6];
    const float* gamma = (const float*)d_in[7];
    float* out = (float*)d_out;

    // workspace: qt (1 MB) | kt (1 MB) | vc (8 MB)  -- all bf16
    unsigned short* qt = (unsigned short*)d_ws;
    unsigned short* kt = qt + (size_t)B_ * N_ * CQK_;
    unsigned short* vc = kt + (size_t)B_ * N_ * CQK_;

    proj_qkv<<<B_ * (N_ / 32), 256, 0, stream>>>(x, wq, bq, wk, bk, wv, bv, qt, kt, vc);
    // 8 waves/block, 128 queries/block -> 64 blocks per batch
    flash_attn<<<B_ * (N_ / 128), 256, 0, stream>>>(qt, kt, vc, x, gamma, out);
}
